// TrittentionCube_2121713844824
// MI455X (gfx1250) — matrix operand order repaired and checked by reading, not compiled, not run on hardware
//
#include <hip/hip_runtime.h>

// ---------------------------------------------------------------------------
// TrittentionCube for MI455X (gfx1250, wave32, WMMA f32 += f16 16x16x32).
//
// v-tensor algebraically eliminated:
//   z[n,r,:] = invZ * W_Vn : (td^T * exp(scores) * te)
// The S = td^T E te matrices (64x64 per (n,r)) are written to global and the
// W_V contraction is a batched WMMA GEMM (no scalar-FMA sweep over W_V).
//
// Single-pass fused attention per (n,r): full 256x256 logits tile in LDS
// (128 KB dynamic shared; CDNA5 allows up to 320 KB/workgroup), causal-dead
// fragments (p>=q or q>=r, ~5/6 of the plane) skipped wave-uniformly.
// Wave indices go through readfirstlane so skip conditions are provably
// uniform -> scalar branches; WMMA always runs with EXEC all-ones (ISA req).
// All WMMA operands are contiguous 16B chunks per lane -> b128 loads:
//   A operands: row-major [M][K];  B operands: transposed row-major [N][K].
// ---------------------------------------------------------------------------

#define DEV __device__ __forceinline__

typedef __attribute__((ext_vector_type(16))) _Float16 v16h;
typedef __attribute__((ext_vector_type(8)))  _Float16 v8h;
typedef __attribute__((ext_vector_type(8)))  float    v8f;

constexpr int P = 256, D = 512, N = 8, H = 64;

DEV int lane_id() { return threadIdx.x & 31; }
DEV int wave_id() { return __builtin_amdgcn_readfirstlane((int)(threadIdx.x >> 5)); }
DEV int imin(int a, int b) { return a < b ? a : b; }
DEV int imax(int a, int b) { return a > b ? a : b; }

// A fragment (16x32 f16) from row-major [M][K]; lane needs K = {8*half+0..7}
// and {16+8*half+0..7}: two contiguous 16B chunks (CDNA5 ISA 7.12.2).
DEV v16h load_frag_a_vec(const _Float16* base, int ld, int m0, int k0) {
  int l = lane_id(), m = l & 15, half = l >> 4;
  const _Float16* p = base + (size_t)(m0 + m) * ld + k0 + 8 * half;
  v8h lo = *(const v8h*)p;
  v8h hi = *(const v8h*)(p + 16);
  return __builtin_shufflevector(lo, hi, 0, 1, 2, 3, 4, 5, 6, 7,
                                 8, 9, 10, 11, 12, 13, 14, 15);
}

// B fragment (32x16 f16) from TRANSPOSED source: row-major [Ncols][K].
DEV v16h load_frag_bt_vec(const _Float16* base, int ld, int k0, int c0) {
  int l = lane_id(), nn = l & 15, half = l >> 4;
  const _Float16* p = base + (size_t)(c0 + nn) * ld + k0 + 8 * half;
  v8h lo = *(const v8h*)p;
  v8h hi = *(const v8h*)(p + 16);
  return __builtin_shufflevector(lo, hi, 0, 1, 2, 3, 4, 5, 6, 7,
                                 8, 9, 10, 11, 12, 13, 14, 15);
}

DEV v8f wmma_f16(v16h a, v16h b, v8f c) {
  return __builtin_amdgcn_wmma_f32_16x16x32_f16(false, a, false, b, (short)0, c, false, false);
}

// ---------------------------------------------------------------------------
// Generic WMMA GEMM: 128 threads = 4 waves; each wave one 16x32 tile.
// grid = (Ncols/32, M/64).  out = (acc + bias) * scale
// ---------------------------------------------------------------------------
template <bool OUTF32, bool BIAS>
__global__ __launch_bounds__(128) void gemm16_kernel(
    const _Float16* __restrict__ A, int lda, const _Float16* __restrict__ Bt, int ldb,
    _Float16* __restrict__ Ch, float* __restrict__ Cf, int ldc,
    const float* __restrict__ bias, float scale, int K) {
  int wave = wave_id();
  int tm = blockIdx.y * 64 + wave * 16;
  int tn = blockIdx.x * 32;
  v8f acc0 = {}, acc1 = {};
  for (int k = 0; k < K; k += 32) {
    if (k + 32 < K) {  // global_prefetch_b8 hint for next K-slab
      __builtin_prefetch(A + (size_t)(tm + (lane_id() & 15)) * lda + k + 32, 0, 3);
      __builtin_prefetch(Bt + (size_t)(tn + (lane_id() & 15)) * ldb + k + 32, 0, 3);
    }
    v16h a  = load_frag_a_vec(A, lda, tm, k);
    v16h b0 = load_frag_bt_vec(Bt, ldb, k, tn);
    v16h b1 = load_frag_bt_vec(Bt, ldb, k, tn + 16);
    acc0 = wmma_f16(a, b0, acc0);
    acc1 = wmma_f16(a, b1, acc1);
  }
  int l = lane_id(), nn = l & 15, half = l >> 4;
#pragma unroll
  for (int v = 0; v < 8; ++v) {
    int m = tm + v + 8 * half;
    float v0 = acc0[v], v1 = acc1[v];
    if (BIAS) { v0 += bias[tn + nn]; v1 += bias[tn + 16 + nn]; }
    v0 *= scale; v1 *= scale;
    if (OUTF32) {
      Cf[(size_t)m * ldc + tn + nn]      = v0;
      Cf[(size_t)m * ldc + tn + 16 + nn] = v1;
    } else {
      Ch[(size_t)m * ldc + tn + nn]      = (_Float16)v0;
      Ch[(size_t)m * ldc + tn + 16 + nn] = (_Float16)v1;
    }
  }
}

// ---------------------------------------------------------------------------
// Prep kernels (layout permutes + f32->f16)
// ---------------------------------------------------------------------------
__global__ void cvt_f32_f16(const float* __restrict__ in, _Float16* __restrict__ out, int n) {
  int i = blockIdx.x * blockDim.x + threadIdx.x;
  if (i < n) out[i] = (_Float16)in[i];
}
// W_s [n][d][h] f32 -> wprojT [(n*H+h)][d] f16
__global__ void permute_wproj(const float* __restrict__ in, _Float16* __restrict__ out) {
  int i = blockIdx.x * blockDim.x + threadIdx.x;
  if (i >= N * D * H) return;
  int h = i % H, d = (i / H) % D, n = i / (H * D);
  out[(size_t)(n * H + h) * D + d] = (_Float16)in[i];
}
// W_K [n][i][j][k] f32 -> wkT [(i*H+j)][(n*H+k)] f16
__global__ void permute_wk(const float* __restrict__ in, _Float16* __restrict__ out) {
  int idx = blockIdx.x * blockDim.x + threadIdx.x;
  if (idx >= (H * H) * (N * H)) return;
  int col = idx % (N * H), row = idx / (N * H);
  int i = row / H, j = row % H;
  int n = col / H, k = col % H;
  out[idx] = (_Float16)in[(((size_t)n * H + i) * H + j) * H + k];
}
// W_O [(n,h)][d] f32 -> wohT [d][(n,h)] f16
__global__ void transpose_wo(const float* __restrict__ in, _Float16* __restrict__ out) {
  int idx = blockIdx.x * blockDim.x + threadIdx.x;
  if (idx >= D * (N * H)) return;
  int d = idx / (N * H), nh = idx % (N * H);
  out[idx] = (_Float16)in[(size_t)nh * D + d];
}
// W_V [n][(h,g)][f] f32 -> wvT [n][f][(h,g)] f16   (B^T for z GEMM)
__global__ void permute_wv(const float* __restrict__ in, _Float16* __restrict__ out) {
  int idx = blockIdx.x * blockDim.x + threadIdx.x;
  if (idx >= N * H * (H * H)) return;
  int n = idx / (H * H * H);
  int rem = idx % (H * H * H);
  int f = rem / (H * H), hg = rem % (H * H);
  out[idx] = (_Float16)in[(size_t)n * H * H * H + (size_t)hg * H + f];
}
// t [q][(n,h)] f16 -> tT [(n,h)][q] f16   (for td, te)
__global__ void transpose_t16(const _Float16* __restrict__ in, _Float16* __restrict__ out) {
  int idx = blockIdx.x * blockDim.x + threadIdx.x;
  if (idx >= (N * H) * P) return;
  int nh = idx / P, q = idx % P;
  out[idx] = in[(size_t)q * (N * H) + nh];
}
// tbsum[q][j] = sum_n tb[q][(n,j)]
__global__ void tbsum_kernel(const _Float16* __restrict__ tb, _Float16* __restrict__ tbs) {
  int i = blockIdx.x * blockDim.x + threadIdx.x;
  if (i >= P * H) return;
  int j = i % H, q = i / H;
  float s = 0.f;
  for (int nn = 0; nn < N; ++nn) s += (float)tb[(size_t)q * (N * H) + nn * H + j];
  tbs[i] = (_Float16)s;
}

// ---------------------------------------------------------------------------
// Fused attention: one workgroup per (r, n). 256 threads = 8 wave32s.
// Dynamic LDS: sc[q][p] 256x256 f16 (128 KB) + Vt[h][q] 64x256 f16 (32 KB).
// Writes S*invZ (64x64 f16) to sbuf[n][r].
// ---------------------------------------------------------------------------
__global__ __launch_bounds__(256) void fused_attn_kernel(
    const _Float16* __restrict__ ta,    // [p][(n,i)]  ld 512, pre-scaled by 1/H
    const _Float16* __restrict__ tdT,   // [(n,h)][p]  ld 256
    const _Float16* __restrict__ teT,   // [(n,g)][q]  ld 256
    const _Float16* __restrict__ st2T,  // [q][(r,i)]  ld 16384
    _Float16* __restrict__ sbuf) {      // [n][r][(h,g)]
  extern __shared__ __align__(16) char dyn[];
  _Float16* sc = (_Float16*)dyn;              // attn tile, [q][p]
  _Float16* vt = (_Float16*)(dyn + 131072);   // Vt = td^T @ E, [h][q]
  __shared__ float red_m[8], red_l[8];

  const int r = blockIdx.x, n = blockIdx.y;
  const int tid = threadIdx.x, l = tid & 31;
  const int wave = wave_id();                 // provably wave-uniform
  const int nn16 = l & 15, half = l >> 4;
  const int p0 = wave * 32;
  const _Float16* s2t = st2T + (size_t)r * H;   // B^T slice [q][i], ld P*H

  // ---- phase 1: masked logits -> sc[q][p]; dead fragments skipped ----
  // The exp sweep only reads locations with p < roundup8(q) of live rows, so
  // dead fragments (all p>=q, or q>=r) need no store at all.
  v16h afr[2][2];
#pragma unroll
  for (int mt = 0; mt < 2; ++mt) {
    afr[mt][0] = load_frag_a_vec(ta, N * H, p0 + mt * 16, n * H);
    afr[mt][1] = load_frag_a_vec(ta, N * H, p0 + mt * 16, n * H + 32);
  }
  float m_t = -1000.0f;   // softmax "extra" column logit
  for (int qb = 0; qb < P; qb += 32) {
    if (qb >= r) break;   // all remaining q >= r: fully masked (uniform)
#pragma unroll
    for (int nt = 0; nt < 2; ++nt) {
      int ql = qb + nt * 16;
      if (ql >= r) continue;                 // uniform
      v16h b0 = load_frag_bt_vec(s2t, P * H, 0, ql);
      v16h b1 = load_frag_bt_vec(s2t, P * H, 32, ql);
      int q = ql + nn16;
#pragma unroll
      for (int mt = 0; mt < 2; ++mt) {
        int pm = p0 + mt * 16;
        if (pm >= ql + 15) continue;         // uniform: all p >= q here
        v8f acc = {};
        acc = wmma_f16(afr[mt][0], b0, acc);
        acc = wmma_f16(afr[mt][1], b1, acc);
        v8h row;
#pragma unroll
        for (int v = 0; v < 8; ++v) {
          int p = pm + v + 8 * half;
          float lg = (p < q && q < r) ? acc[v] : -15625.0f;  // IGNORE/H
          m_t = fmaxf(m_t, lg);
          row[v] = (_Float16)lg;
        }
        *(v8h*)&sc[(size_t)q * P + pm + 8 * half] = row;  // ds b128
      }
    }
  }
#pragma unroll
  for (int o = 16; o > 0; o >>= 1) m_t = fmaxf(m_t, __shfl_xor(m_t, o, 32));
  if (l == 0) red_m[wave] = m_t;
  __syncthreads();
  float m_g = red_m[0];
#pragma unroll
  for (int i = 1; i < 8; ++i) m_g = fmaxf(m_g, red_m[i]);

  // ---- phase 2: exp sweep of row q = tid; zero-fill the masked region ----
  float l_t = 0.f;
  {
    int qv = (tid == 0 || tid >= r) ? 0 : tid;   // valid columns: p < q
    _Float16* rowp = sc + (size_t)tid * P;
    v8h zrow = {};
    for (int c = 0; c < P; c += 8) {
      if (c < qv) {
        v8h hv = *(v8h*)&rowp[c];
        v8h ev;
#pragma unroll
        for (int e8 = 0; e8 < 8; ++e8) {
          float e = __expf((float)hv[e8] - m_g);
          l_t += e;
          ev[e8] = (_Float16)e;
        }
        *(v8h*)&rowp[c] = ev;
      } else {
        *(v8h*)&rowp[c] = zrow;
      }
    }
  }
#pragma unroll
  for (int o = 16; o > 0; o >>= 1) l_t += __shfl_xor(l_t, o, 32);
  if (l == 0) red_l[wave] = l_t;
  __syncthreads();
  float Z = __expf(-1000.0f - m_g);   // extra column
#pragma unroll
  for (int i = 0; i < 8; ++i) Z += red_l[i];
  float invZ = 1.0f / Z;

  // ---- phase 3: Vt[h][q] = td_n^T @ E  (wave: one ht, 8 q-tiles; shared A) --
  {
    int ht = wave >> 1, qt0 = (wave & 1) * 8;
    v8f acc[8];
#pragma unroll
    for (int t = 0; t < 8; ++t) acc[t] = (v8f){};
    int kend[8], kmaxw = 0;
#pragma unroll
    for (int t = 0; t < 8; ++t) {
      int qt = qt0 + t;
      int ke = (qt * 16 >= r) ? 0 : imin(qt * 16 + 16, r);  // e=0 for p>=q or q>=r
      ke = (ke + 31) & ~31;
      kend[t] = ke;
      kmaxw = imax(kmaxw, ke);
    }
    for (int k = 0; k < kmaxw; k += 32) {
      v16h a = load_frag_a_vec(tdT + (size_t)n * H * P, P, ht * 16, k);
#pragma unroll
      for (int t = 0; t < 8; ++t)
        if (k < kend[t])   // uniform (ht/qt0 from readfirstlane'd wave)
          acc[t] = wmma_f16(a, load_frag_bt_vec(sc, P, k, (qt0 + t) * 16), acc[t]);
    }
#pragma unroll
    for (int t = 0; t < 8; ++t)
#pragma unroll
      for (int v = 0; v < 8; ++v)
        vt[(size_t)(ht * 16 + v + 8 * half) * P + (qt0 + t) * 16 + nn16] =
            (_Float16)acc[t][v];
  }
  __syncthreads();

  // ---- phase 4: S = Vt @ te_n; write S*invZ (f16) to sbuf[n][r] ----
  {
    int ht = wave >> 1, gt0 = (wave & 1) * 2;
    int kend = (r >= 2) ? ((imin(P, r) + 31) & ~31) : 0;  // Vt[:,q>=r] == 0
    v8f s0 = {}, s1 = {};
    for (int k = 0; k < kend; k += 32) {
      v16h a = load_frag_a_vec(vt, P, ht * 16, k);
      s0 = wmma_f16(a, load_frag_bt_vec(teT + (size_t)n * H * P, P, k, gt0 * 16), s0);
      s1 = wmma_f16(a, load_frag_bt_vec(teT + (size_t)n * H * P, P, k, gt0 * 16 + 16), s1);
    }
    _Float16* sb = sbuf + ((size_t)n * P + r) * (H * H);
#pragma unroll
    for (int v = 0; v < 8; ++v) {
      int hh = ht * 16 + v + 8 * half;
      sb[hh * H + gt0 * 16 + nn16]      = (_Float16)(s0[v] * invZ);
      sb[hh * H + gt0 * 16 + 16 + nn16] = (_Float16)(s1[v] * invZ);
    }
  }
}

// ---------------------------------------------------------------------------
extern "C" void kernel_launch(void* const* d_in, const int* in_sizes, int n_in,
                              void* d_out, int out_size, void* d_ws, size_t ws_size,
                              hipStream_t stream) {
  (void)in_sizes; (void)n_in; (void)out_size; (void)ws_size;
  const float* x = (const float*)d_in[0];
  const float* Wp[5] = {(const float*)d_in[1], (const float*)d_in[3], (const float*)d_in[5],
                        (const float*)d_in[7], (const float*)d_in[9]};
  const float* bp[5] = {(const float*)d_in[2], (const float*)d_in[4], (const float*)d_in[6],
                        (const float*)d_in[8], (const float*)d_in[10]};
  const float* W_V = (const float*)d_in[11];
  const float* W_K = (const float*)d_in[12];
  const float* W_O = (const float*)d_in[13];
  const float* b_O = (const float*)d_in[14];
  float* out = (float*)d_out;

  char* ws = (char*)d_ws;
  size_t off = 0;
  auto alloc = [&](size_t bytes) -> void* {
    void* p = ws + off;
    off += (bytes + 255) & ~(size_t)255;
    return p;
  };
  _Float16* xh     = (_Float16*)alloc((size_t)P * D * 2);
  _Float16* wprojT = (_Float16*)alloc((size_t)5 * (N * H) * D * 2);
  _Float16* tbuf   = (_Float16*)alloc((size_t)5 * P * (N * H) * 2);
  _Float16* tdT    = (_Float16*)alloc((size_t)(N * H) * P * 2);
  _Float16* teT    = (_Float16*)alloc((size_t)(N * H) * P * 2);
  _Float16* tbs    = (_Float16*)alloc((size_t)P * H * 2);
  _Float16* wkT    = (_Float16*)alloc((size_t)(H * H) * (N * H) * 2);
  _Float16* st1    = (_Float16*)alloc((size_t)P * (H * H) * 2);
  _Float16* st2T   = (_Float16*)alloc((size_t)P * (P * H) * 2);
  _Float16* zb     = (_Float16*)alloc((size_t)P * (N * H) * 2);
  _Float16* wohT   = (_Float16*)alloc((size_t)D * (N * H) * 2);
  _Float16* sbuf   = (_Float16*)alloc((size_t)N * P * (H * H) * 2);   // 16 MB
  _Float16* wvT    = (_Float16*)alloc((size_t)N * H * (H * H) * 2);   // 4 MB

  _Float16* ta = tbuf + (size_t)0 * P * (N * H);
  _Float16* tb = tbuf + (size_t)1 * P * (N * H);
  _Float16* tc = tbuf + (size_t)2 * P * (N * H);
  _Float16* td = tbuf + (size_t)3 * P * (N * H);
  _Float16* te = tbuf + (size_t)4 * P * (N * H);

  // operand prep
  cvt_f32_f16<<<(P * D + 255) / 256, 256, 0, stream>>>(x, xh, P * D);
  for (int s = 0; s < 5; ++s)
    permute_wproj<<<(N * D * H + 255) / 256, 256, 0, stream>>>(
        Wp[s], wprojT + (size_t)s * (N * H) * D);
  permute_wk<<<((H * H) * (N * H) + 255) / 256, 256, 0, stream>>>(W_K, wkT);
  transpose_wo<<<(D * (N * H) + 255) / 256, 256, 0, stream>>>(W_O, wohT);
  permute_wv<<<(N * H * H * H + 255) / 256, 256, 0, stream>>>(W_V, wvT);

  // projections t_s = (x @ W_s + b_s) * scale; letter A folds the 1/H scale.
  for (int s = 0; s < 5; ++s)
    gemm16_kernel<false, true><<<dim3((N * H) / 32, P / 64), 128, 0, stream>>>(
        xh, D, wprojT + (size_t)s * (N * H) * D, D,
        tbuf + (size_t)s * P * (N * H), nullptr, N * H, bp[s],
        (s == 0) ? (1.0f / H) : 1.0f, D);

  tbsum_kernel<<<(P * H + 255) / 256, 256, 0, stream>>>(tb, tbs);
  transpose_t16<<<((N * H) * P + 255) / 256, 256, 0, stream>>>(td, tdT);
  transpose_t16<<<((N * H) * P + 255) / 256, 256, 0, stream>>>(te, teT);

  // step1[r,(i,j)] = tc[r,(n,k)] @ W_K^T   (M=256, N=4096, K=512)
  gemm16_kernel<false, false><<<dim3((H * H) / 32, P / 64), 128, 0, stream>>>(
      tc, N * H, wkT, N * H, st1, nullptr, H * H, nullptr, 1.0f, N * H);

  // step2^T[q,(r,i)] = tbsum[q,j] @ step1[(r,i),j]^T   (M=256, N=16384, K=64)
  gemm16_kernel<false, false><<<dim3((P * H) / 32, P / 64), 128, 0, stream>>>(
      tbs, H, st1, H, st2T, nullptr, P * H, nullptr, 1.0f, H);

  // fused scores -> global softmax -> S*invZ  per (r, n); 160 KB dynamic LDS
  fused_attn_kernel<<<dim3(P, N), 256, 160 * 1024, stream>>>(ta, tdT, teT, st2T, sbuf);

  // z[r,(n,f)] = S_n[r,(h,g)] @ W_Vn[(h,g),f]  (8x GEMM: M=256, N=64, K=4096)
  for (int nn = 0; nn < N; ++nn)
    gemm16_kernel<false, false><<<dim3(H / 32, P / 64), 128, 0, stream>>>(
        sbuf + (size_t)nn * P * (H * H), H * H, wvT + (size_t)nn * H * (H * H), H * H,
        zb + nn * H, nullptr, N * H, nullptr, 1.0f, H * H);

  // out[r,d] = z[r,(n,h)] @ W_O + b_O   (M=256, N=512, K=512)
  gemm16_kernel<true, true><<<dim3(D / 32, P / 64), 128, 0, stream>>>(
      zb, N * H, wohT, D, nullptr, out, D, b_O, 1.0f, N * H);
}
